// TokenSwapMamba_25168508354815
// MI455X (gfx1250) — matrix operand order repaired and checked by reading, not compiled, not run on hardware
//
#include <hip/hip_runtime.h>
#include <hip/hip_bf16.h>

// ---------------- problem constants (fixed by the reference) ----------------
#define B_SZ   2
#define N_TOK  4096
#define DIM    128
#define D_IN   256
#define D2     512          // 2*D_IN
#define D_ST   16
#define DT_RK  8
#define NXP    40           // DT_R + 2*D_ST
#define CHUNK  64
#define NCH    (N_TOK / CHUNK)
#define BN     (B_SZ * N_TOK)   // 8192 rows
#define EPSV   1e-5f

typedef __attribute__((ext_vector_type(16))) __bf16 v16bf;
typedef __attribute__((ext_vector_type(8)))  __bf16 v8bf;
typedef __attribute__((ext_vector_type(8)))  float  v8f;

union ABfrag { v16bf v; v8bf h[2]; };

__device__ __forceinline__ float silu_f(float x)     { return x * (1.0f / (1.0f + __expf(-x))); }
__device__ __forceinline__ float softplus_f(float x) { return (x > 20.0f) ? x : __logf(1.0f + __expf(x)); }

// ---------------- weight convert + transpose: (R,C) f32 -> (C,R) bf16 ------
__global__ void transpose_to_bf16(const float* __restrict__ src,
                                  __bf16* __restrict__ dst, int R, int C) {
  int idx = blockIdx.x * blockDim.x + threadIdx.x;
  if (idx < R * C) {
    int r = idx / C, c = idx % C;
    dst[(size_t)c * R + r] = (__bf16)src[idx];
  }
}

// ---------------- fused residual + LN + channel swap -> bf16 activations ---
__device__ __forceinline__ float block_sum128(float v, float* red) {
  int t = threadIdx.x;
  red[t] = v; __syncthreads();
  #pragma unroll
  for (int off = DIM / 2; off > 0; off >>= 1) {
    if (t < off) red[t] += red[t + off];
    __syncthreads();
  }
  float r = red[0]; __syncthreads();
  return r;
}

__global__ __launch_bounds__(DIM)
void ln_swap_kernel(const float* __restrict__ I1, const float* __restrict__ I2,
                    const float* __restrict__ R1, const float* __restrict__ R2,
                    const float* __restrict__ w1, const float* __restrict__ b1,
                    const float* __restrict__ w2, const float* __restrict__ b2,
                    float* __restrict__ outR1, float* __restrict__ outR2,
                    __bf16* __restrict__ x1s, __bf16* __restrict__ x2s) {
  int row = blockIdx.x;               // 0..BN-1
  int t   = threadIdx.x;              // channel 0..127
  size_t idx = (size_t)row * DIM + t;
  float s1 = I1[idx] + R1[idx];
  float s2 = I2[idx] + R2[idx];
  outR1[idx] = s1; outR2[idx] = s2;

  __shared__ float red[DIM];
  float mu1 = block_sum128(s1,      red) * (1.0f / DIM);
  float sq1 = block_sum128(s1 * s1, red) * (1.0f / DIM);
  float mu2 = block_sum128(s2,      red) * (1.0f / DIM);
  float sq2 = block_sum128(s2 * s2, red) * (1.0f / DIM);
  float n1 = (s1 - mu1) * rsqrtf(sq1 - mu1 * mu1 + EPSV) * w1[t] + b1[t];
  float n2 = (s2 - mu2) * rsqrtf(sq2 - mu2 * mu2 + EPSV) * w2[t] + b2[t];
  bool even = (t & 1) == 0;                      // even channels swap streams
  x1s[idx] = (__bf16)(even ? n2 : n1);
  x2s[idx] = (__bf16)(even ? n1 : n2);
}

// ---------------- bf16 WMMA GEMM: C(M,N) = A(M,K) * Bt(N,K)^T --------------
// A row-major bf16, Bt row-major bf16 (pre-transposed weights), C row-major f32.
// Wave tile: 16(M) x 64(N), K stepped by 32 with v_wmma_f32_16x16x32_bf16.
__global__ __launch_bounds__(256)
void gemm_bf16(const __bf16* __restrict__ A, const __bf16* __restrict__ Bt,
               float* __restrict__ C, int M, int Nn, int K) {
  int wave   = blockIdx.x * (blockDim.x >> 5) + (threadIdx.x >> 5);
  int lane   = threadIdx.x & 31;
  int nlane  = lane & 15;
  int kh     = lane >> 4;             // K-half selector per ISA A/B layouts
  int ntiles = Nn >> 6;
  int mt = wave / ntiles;
  int nt = wave % ntiles;
  if (mt * 16 >= M) return;           // wave-uniform guard (EXEC stays all-1)

  v8f acc[4] = {v8f{}, v8f{}, v8f{}, v8f{}};
  const __bf16* arow = A + (size_t)(mt * 16 + nlane) * K;

  for (int k0 = 0; k0 < K; k0 += 32) {
    // A fragment 16x32: lanes 0-15 hold K {0..7,16..23}, lanes 16-31 {8..15,24..31}
    ABfrag a;
    a.h[0] = *(const v8bf*)(arow + k0 + kh * 8);
    a.h[1] = *(const v8bf*)(arow + k0 + 16 + kh * 8);
    __builtin_prefetch(arow + k0 + 64, 0, 1);   // global_prefetch next K panel
    #pragma unroll
    for (int t = 0; t < 4; ++t) {
      // B fragment 32x16: lane holds column n, K {kh*8..kh*8+7, 16+kh*8..16+kh*8+7} in Bt
      const __bf16* brow = Bt + (size_t)(nt * 64 + t * 16 + nlane) * K + k0 + kh * 8;
      ABfrag b;
      b.h[0] = *(const v8bf*)(brow);
      b.h[1] = *(const v8bf*)(brow + 16);
      acc[t] = __builtin_amdgcn_wmma_f32_16x16x32_bf16(
          /*neg_a=*/false, a.v, /*neg_b=*/false, b.v,
          /*c_mod=*/(short)0, acc[t], /*reuse_a=*/false, /*reuse_b=*/false);
    }
  }
  // D layout: VGPR r holds M = mt*16 + r + 8*kh, N = nt*64 + t*16 + nlane
  #pragma unroll
  for (int t = 0; t < 4; ++t) {
    #pragma unroll
    for (int r = 0; r < 8; ++r) {
      int mm = mt * 16 + r + 8 * kh;
      int nn = nt * 64 + t * 16 + nlane;
      C[(size_t)mm * Nn + nn] = acc[t][r];
    }
  }
}

// ---------------- depthwise conv + SiLU + x_dbl + dt -----------------------
__global__ __launch_bounds__(D_IN)
void conv_xdbl_dt(const float* __restrict__ xz,     // (BN, 512): [0,256)=xp, [256,512)=z
                  const float* __restrict__ conv_w, // (256,4)
                  const float* __restrict__ conv_b, // (256)
                  const float* __restrict__ W_x,    // (256,40)
                  const float* __restrict__ W_dt,   // (8,256)
                  const float* __restrict__ dt_bias,// (256)
                  float* __restrict__ xc_out,       // (BN,256)
                  float* __restrict__ dt_out,       // (BN,256)
                  float* __restrict__ Bm_out,       // (BN,16)
                  float* __restrict__ Cm_out) {     // (BN,16)
  int tok = blockIdx.x;       // 0..BN-1
  int l   = tok & (N_TOK - 1);
  int d   = threadIdx.x;      // channel 0..255
  float acc = conv_b[d];
  #pragma unroll
  for (int t = 0; t < 4; ++t) {
    int ll = l - 3 + t;       // causal pad, does not cross batch boundary
    float v = (ll >= 0) ? xz[(size_t)(tok - 3 + t) * D2 + d] : 0.0f;
    acc += conv_w[d * 4 + t] * v;
  }
  float xcv = silu_f(acc);
  __shared__ float sxc[D_IN];
  __shared__ float sdbl[NXP];
  sxc[d] = xcv;
  xc_out[(size_t)tok * D_IN + d] = xcv;
  __syncthreads();
  if (d < NXP) {              // x_dbl = xc @ W_x  (256 -> 40)
    float s = 0.0f;
    for (int k = 0; k < D_IN; ++k) s += sxc[k] * W_x[k * NXP + d];
    sdbl[d] = s;
  }
  __syncthreads();
  float ta = dt_bias[d];      // dt = softplus(dt_r @ W_dt + bias)  (8 -> 256)
  #pragma unroll
  for (int r = 0; r < DT_RK; ++r) ta += sdbl[r] * W_dt[r * D_IN + d];
  dt_out[(size_t)tok * D_IN + d] = softplus_f(ta);
  if (d < D_ST)            Bm_out[(size_t)tok * D_ST + d]        = sdbl[DT_RK + d];
  else if (d < 2 * D_ST)   Cm_out[(size_t)tok * D_ST + d - D_ST] = sdbl[DT_RK + D_ST + (d - D_ST)];
}

// ---------------- chunked linear-recurrence scan ---------------------------
// h_l = dA_l * h_{l-1} + dt_l*B_l*xc_l ; pass1: per-chunk (P = prod dA, c = h|h0=0)
__global__ __launch_bounds__(D_IN)
void scan_pass1(const float* __restrict__ dt, const float* __restrict__ xc,
                const float* __restrict__ Bm, const float* __restrict__ A_log,
                float* __restrict__ chP, float* __restrict__ chH) {
  int bid = blockIdx.x;             // b*NCH + chunk
  int b   = bid / NCH, ch = bid % NCH;
  int d   = threadIdx.x;
  float A[D_ST], h[D_ST], P[D_ST];
  #pragma unroll
  for (int s = 0; s < D_ST; ++s) {
    A[s] = -__expf(A_log[d * D_ST + s]);
    h[s] = 0.0f; P[s] = 1.0f;
  }
  __shared__ float sB[D_ST];
  int base = b * N_TOK + ch * CHUNK;
  for (int i = 0; i < CHUNK; ++i) {
    int tok = base + i;
    if (d < D_ST) sB[d] = Bm[(size_t)tok * D_ST + d];
    __syncthreads();
    float dtv = dt[(size_t)tok * D_IN + d];
    float dx  = dtv * xc[(size_t)tok * D_IN + d];
    #pragma unroll
    for (int s = 0; s < D_ST; ++s) {
      float dA = __expf(dtv * A[s]);
      h[s] = dA * h[s] + dx * sB[s];
      P[s] *= dA;
    }
    __syncthreads();
  }
  size_t o = ((size_t)bid * D_IN + d) * D_ST;
  #pragma unroll
  for (int s = 0; s < D_ST; ++s) { chP[o + s] = P[s]; chH[o + s] = h[s]; }
}

// pass2: serial scan over 64 chunk states per (b) -> exclusive h0 per chunk
__global__ __launch_bounds__(D_IN)
void scan_pass2(const float* __restrict__ chP, const float* __restrict__ chH,
                float* __restrict__ chH0) {
  int b = blockIdx.x, d = threadIdx.x;
  float h0[D_ST];
  #pragma unroll
  for (int s = 0; s < D_ST; ++s) h0[s] = 0.0f;
  for (int ch = 0; ch < NCH; ++ch) {
    size_t o = ((size_t)(b * NCH + ch) * D_IN + d) * D_ST;
    #pragma unroll
    for (int s = 0; s < D_ST; ++s) {
      chH0[o + s] = h0[s];
      h0[s] = chP[o + s] * h0[s] + chH[o + s];
    }
  }
}

// pass3: replay chunk from corrected h0, emit y_l = <h_l, C_l>
__global__ __launch_bounds__(D_IN)
void scan_pass3(const float* __restrict__ dt, const float* __restrict__ xc,
                const float* __restrict__ Bm, const float* __restrict__ Cm,
                const float* __restrict__ A_log, const float* __restrict__ chH0,
                float* __restrict__ ys) {
  int bid = blockIdx.x;
  int b   = bid / NCH, ch = bid % NCH;
  int d   = threadIdx.x;
  float A[D_ST], h[D_ST];
  size_t o = ((size_t)bid * D_IN + d) * D_ST;
  #pragma unroll
  for (int s = 0; s < D_ST; ++s) {
    A[s] = -__expf(A_log[d * D_ST + s]);
    h[s] = chH0[o + s];
  }
  __shared__ float sB[D_ST];
  __shared__ float sC[D_ST];
  int base = b * N_TOK + ch * CHUNK;
  for (int i = 0; i < CHUNK; ++i) {
    int tok = base + i;
    if (d < D_ST)          sB[d]        = Bm[(size_t)tok * D_ST + d];
    else if (d < 2 * D_ST) sC[d - D_ST] = Cm[(size_t)tok * D_ST + (d - D_ST)];
    __syncthreads();
    float dtv = dt[(size_t)tok * D_IN + d];
    float dx  = dtv * xc[(size_t)tok * D_IN + d];
    float y = 0.0f;
    #pragma unroll
    for (int s = 0; s < D_ST; ++s) {
      float dA = __expf(dtv * A[s]);
      h[s] = dA * h[s] + dx * sB[s];
      y += h[s] * sC[s];
    }
    ys[(size_t)tok * D_IN + d] = y;
    __syncthreads();
  }
}

// ---------------- gating: y = (ys + xc*D) * silu(z) -> bf16 ----------------
__global__ __launch_bounds__(256)
void gate_kernel(const float* __restrict__ ys, const float* __restrict__ xc,
                 const float* __restrict__ xz, const float* __restrict__ Dp,
                 __bf16* __restrict__ ybf) {
  size_t idx = (size_t)blockIdx.x * blockDim.x + threadIdx.x;
  if (idx >= (size_t)BN * D_IN) return;
  size_t tok = idx / D_IN;
  int    d   = (int)(idx % D_IN);
  float z = xz[tok * D2 + D_IN + d];
  float y = (ys[idx] + xc[idx] * Dp[d]) * silu_f(z);
  ybf[idx] = (__bf16)y;
}

// ---------------------------------------------------------------------------
extern "C" void kernel_launch(void* const* d_in, const int* in_sizes, int n_in,
                              void* d_out, int out_size, void* d_ws, size_t ws_size,
                              hipStream_t stream) {
  (void)in_sizes; (void)n_in; (void)out_size; (void)ws_size;
  const float* I1  = (const float*)d_in[0];
  const float* I2  = (const float*)d_in[1];
  const float* R1  = (const float*)d_in[2];
  const float* R2  = (const float*)d_in[3];
  const float* ln1w = (const float*)d_in[4];
  const float* ln1b = (const float*)d_in[5];
  const float* ln2w = (const float*)d_in[6];
  const float* ln2b = (const float*)d_in[7];
  const float* W_in[2]    = {(const float*)d_in[8],  (const float*)d_in[17]};
  const float* conv_w[2]  = {(const float*)d_in[9],  (const float*)d_in[18]};
  const float* conv_b[2]  = {(const float*)d_in[10], (const float*)d_in[19]};
  const float* W_x[2]     = {(const float*)d_in[11], (const float*)d_in[20]};
  const float* W_dt[2]    = {(const float*)d_in[12], (const float*)d_in[21]};
  const float* dt_bias[2] = {(const float*)d_in[13], (const float*)d_in[22]};
  const float* A_log[2]   = {(const float*)d_in[14], (const float*)d_in[23]};
  const float* Dp[2]      = {(const float*)d_in[15], (const float*)d_in[24]};
  const float* W_out[2]   = {(const float*)d_in[16], (const float*)d_in[25]};

  float* out    = (float*)d_out;
  float* out1   = out;
  float* out2   = out + (size_t)BN * DIM;
  float* outR1  = out + 2 * (size_t)BN * DIM;
  float* outR2  = out + 3 * (size_t)BN * DIM;

  // ---- workspace carve-up (~106 MB; intermediates live in 192 MB L2) ----
  char*  ws  = (char*)d_ws;
  size_t cur = 0;
  auto alloc = [&](size_t bytes) -> char* {
    cur = (cur + 255) & ~(size_t)255;
    char* p = ws + cur; cur += bytes; return p;
  };
  struct SB {
    __bf16 *xbf, *WinT, *WoutT, *ybf;
    float *xz, *xc, *dt, *Bm, *Cm, *ys, *chP, *chH, *chH0;
  } sb[2];
  for (int s = 0; s < 2; ++s) {
    sb[s].xbf   = (__bf16*)alloc((size_t)BN * DIM * 2);
    sb[s].WinT  = (__bf16*)alloc((size_t)D2 * DIM * 2);
    sb[s].WoutT = (__bf16*)alloc((size_t)DIM * D_IN * 2);
    sb[s].ybf   = (__bf16*)alloc((size_t)BN * D_IN * 2);
    sb[s].xz    = (float*)alloc((size_t)BN * D2 * 4);
    sb[s].xc    = (float*)alloc((size_t)BN * D_IN * 4);
    sb[s].dt    = (float*)alloc((size_t)BN * D_IN * 4);
    sb[s].Bm    = (float*)alloc((size_t)BN * D_ST * 4);
    sb[s].Cm    = (float*)alloc((size_t)BN * D_ST * 4);
    sb[s].ys    = (float*)alloc((size_t)BN * D_IN * 4);
    size_t chb  = (size_t)B_SZ * NCH * D_IN * D_ST * 4;
    sb[s].chP   = (float*)alloc(chb);
    sb[s].chH   = (float*)alloc(chb);
    sb[s].chH0  = (float*)alloc(chb);
  }

  // 1) weights -> transposed bf16
  for (int s = 0; s < 2; ++s) {
    transpose_to_bf16<<<(DIM * D2 + 255) / 256, 256, 0, stream>>>(W_in[s],  sb[s].WinT,  DIM,  D2);
    transpose_to_bf16<<<(D_IN * DIM + 255) / 256, 256, 0, stream>>>(W_out[s], sb[s].WoutT, D_IN, DIM);
  }

  // 2) residual + LN + swap -> bf16 activations, residual outputs
  ln_swap_kernel<<<BN, DIM, 0, stream>>>(I1, I2, R1, R2, ln1w, ln1b, ln2w, ln2b,
                                         outR1, outR2, sb[0].xbf, sb[1].xbf);

  for (int s = 0; s < 2; ++s) {
    // 3) xz = x @ W_in  (8192x128 * 128x512), bf16 WMMA
    int waves_xz = (BN / 16) * (D2 / 64);
    gemm_bf16<<<waves_xz / 8, 256, 0, stream>>>(sb[s].xbf, sb[s].WinT, sb[s].xz, BN, D2, DIM);

    // 4) conv + SiLU + x_dbl + dt
    conv_xdbl_dt<<<BN, D_IN, 0, stream>>>(sb[s].xz, conv_w[s], conv_b[s], W_x[s],
                                          W_dt[s], dt_bias[s],
                                          sb[s].xc, sb[s].dt, sb[s].Bm, sb[s].Cm);

    // 5) chunked selective scan
    scan_pass1<<<B_SZ * NCH, D_IN, 0, stream>>>(sb[s].dt, sb[s].xc, sb[s].Bm, A_log[s],
                                                sb[s].chP, sb[s].chH);
    scan_pass2<<<B_SZ, D_IN, 0, stream>>>(sb[s].chP, sb[s].chH, sb[s].chH0);
    scan_pass3<<<B_SZ * NCH, D_IN, 0, stream>>>(sb[s].dt, sb[s].xc, sb[s].Bm, sb[s].Cm,
                                                A_log[s], sb[s].chH0, sb[s].ys);

    // 6) gating -> bf16
    gate_kernel<<<(BN * D_IN + 255) / 256, 256, 0, stream>>>(sb[s].ys, sb[s].xc, sb[s].xz,
                                                             Dp[s], sb[s].ybf);

    // 7) out = y @ W_out  (8192x256 * 256x128), bf16 WMMA, straight to d_out
    int waves_out = (BN / 16) * (DIM / 64);
    gemm_bf16<<<waves_out / 8, 256, 0, stream>>>(sb[s].ybf, sb[s].WoutT,
                                                 (s == 0) ? out1 : out2, BN, DIM, D_IN);
  }
}
